// GraphEncoder_15487652069881
// MI455X (gfx1250) — matrix operand order repaired and checked by reading, not compiled, not run on hardware
//
#include <hip/hip_runtime.h>

typedef __attribute__((ext_vector_type(16))) __bf16 v16bf;
typedef __attribute__((ext_vector_type(8)))  __bf16 v8bf;
typedef __attribute__((ext_vector_type(4)))  __bf16 v4bf;
typedef __attribute__((ext_vector_type(8)))  float  v8f;

#define Bn   16
#define Nn   127
#define Mn   128
#define Dn   64
#define Hn   8

// ---------- WMMA helpers (CDNA5 16x16x32 bf16, wave32) ----------

__device__ __forceinline__ v8f wmma_bf16(v16bf a, v16bf b, v8f c) {
  return __builtin_amdgcn_wmma_f32_16x16x32_bf16(false, a, false, b, (short)0, c,
                                                 false, false);
}

__device__ __forceinline__ v16bf combine(v8bf lo, v8bf hi) {
  return __builtin_shufflevector(lo, hi, 0, 1, 2, 3, 4, 5, 6, 7, 8, 9, 10, 11,
                                 12, 13, 14, 15);
}

// A fragment (row-major A, 16x32 tile): element e -> k = half*16+e
// => two contiguous 16B LDS loads.
__device__ __forceinline__ v16bf afrag(const __bf16* p, int stride, int row0,
                                       int k0, int lane) {
  int row = row0 + (lane & 15);
  int hf  = lane >> 4;
  const __bf16* base = p + row * stride + k0 + hf * 16;
  v8bf lo = *(const v8bf*)base;
  v8bf hi = *(const v8bf*)(base + 8);
  return combine(lo, hi);
}

// B fragment from TRANSPOSED B (Bt[col][k]): element e -> k = half*16+e
// => one contiguous 32B run per lane.
__device__ __forceinline__ v16bf bfragT(const __bf16* p, int stride, int k0,
                                        int col0, int lane) {
  int col = col0 + (lane & 15);
  int hf  = lane >> 4;
  const __bf16* base = p + col * stride + k0 + hf * 16;
  v8bf lo = *(const v8bf*)base;
  v8bf hi = *(const v8bf*)(base + 8);
  return combine(lo, hi);
}

__device__ __forceinline__ v8bf pack_bf8(v8f a) {
  v8bf r;
#pragma unroll
  for (int e = 0; e < 8; ++e) r[e] = (__bf16)a[e];
  return r;
}

// ---------- Kernel 1: build x0 = LN(concat(ef,ni,nj) @ W_in + b_in) ----------
__global__ __launch_bounds__(256) void build_kernel(
    const float* __restrict__ nodef, const float* __restrict__ edgef,
    const float* __restrict__ W_in, const float* __restrict__ b_in,
    const float* __restrict__ lng, const float* __restrict__ lnb,
    float* __restrict__ X) {
  int wid = threadIdx.x >> 5, lane = threadIdx.x & 31;
  int t   = blockIdx.x * 8 + wid;  // token index, t < 16*128*128
  int b   = t >> 14;
  int rem = t & 16383;
  int i   = rem >> 7;
  int j   = rem & 127;

  float f[43];
#pragma unroll
  for (int e = 0; e < 8; ++e)
    f[e] = (i > 0 && j > 0)
               ? edgef[(((b * Nn + (i - 1)) * Nn) + (j - 1)) * 8 + e]
               : 0.0f;
  f[8] = (i == 0 || j == 0) ? 1.0f : 0.0f;
#pragma unroll
  for (int c = 0; c < 16; ++c)
    f[9 + c] = (i > 0) ? nodef[(b * Nn + (i - 1)) * 16 + c] : 0.0f;
  f[25] = 0.0f;
#pragma unroll
  for (int c = 0; c < 16; ++c)
    f[26 + c] = (j > 0) ? nodef[(b * Nn + (j - 1)) * 16 + c] : 0.0f;
  f[42] = 0.0f;

  float y0 = b_in[lane], y1 = b_in[lane + 32];
  for (int c = 0; c < 43; ++c) {
    y0 += f[c] * W_in[c * 64 + lane];
    y1 += f[c] * W_in[c * 64 + lane + 32];
  }
  float s = y0 + y1;
  s += __shfl_xor(s, 1);  s += __shfl_xor(s, 2);  s += __shfl_xor(s, 4);
  s += __shfl_xor(s, 8);  s += __shfl_xor(s, 16);
  float mean = s * (1.0f / 64.0f);
  float d0 = y0 - mean, d1 = y1 - mean;
  float sq = d0 * d0 + d1 * d1;
  sq += __shfl_xor(sq, 1); sq += __shfl_xor(sq, 2); sq += __shfl_xor(sq, 4);
  sq += __shfl_xor(sq, 8); sq += __shfl_xor(sq, 16);
  float inv = rsqrtf(sq * (1.0f / 64.0f) + 1e-5f);
  float* row = X + (size_t)t * 64;
  row[lane]      = d0 * inv * lng[lane] + lnb[lane];
  row[lane + 32] = d1 * inv * lng[lane + 32] + lnb[lane + 32];
}

// ---------- Kernel 2: one fused transformer layer for one (b,i) row ----------
// block = 256 threads = 8 waves; wave w owns token/query strip [16w, 16w+16)
__global__ __launch_bounds__(256) void layer_kernel(
    float* __restrict__ X, const float* __restrict__ mask,
    const float* __restrict__ Wq, const float* __restrict__ Wk,
    const float* __restrict__ Wv, const float* __restrict__ Wo,
    const float* __restrict__ bo, const float* __restrict__ W1,
    const float* __restrict__ b1, const float* __restrict__ W2,
    const float* __restrict__ b2, const float* __restrict__ ln1g,
    const float* __restrict__ ln1b, const float* __restrict__ ln2g,
    const float* __restrict__ ln2b) {
  __shared__ float  sXf[Mn * Dn];   // fp32 activations (residual/LN)
  __shared__ float  sNeg[Mn];       // key-mask additive term
  __shared__ __bf16 sXh[Mn * Dn];   // bf16 activations; reused for attn out O
  __shared__ __bf16 sQ[Mn * Dn];    // [query][hk] row-major (Bt for K.Q^T)
  __shared__ __bf16 sK[Mn * Dn];    // [key][hk]   row-major (A  for K.Q^T)
  __shared__ __bf16 sVt[Dn * Mn];   // [hv][key]   (Bt for P.V)
  __shared__ __bf16 sP[Mn * Mn];    // [query][key] row-major (A for P.V)
  __shared__ __bf16 sW[12288];      // transposed weight staging / FFN scratch

  const int tid  = threadIdx.x;
  const int lane = tid & 31, w = tid >> 5;
  const int hf = lane >> 4, l15 = lane & 15;
  const int bi = blockIdx.x;
  const int b  = bi >> 7, i = bi & 127;
  float* Xrow = X + (size_t)bi * Mn * Dn;

  // ---- stage row (vectorized) + mask + transposed QKV weights ----
  {
    const float4* Xv = (const float4*)Xrow;
    float4* sXv = (float4*)sXf;
    v4bf* sHv = (v4bf*)sXh;
    for (int idx = tid; idx < 2048; idx += 256) {
      float4 v = Xv[idx];
      sXv[idx] = v;
      v4bf pk;
      pk[0] = (__bf16)v.x; pk[1] = (__bf16)v.y;
      pk[2] = (__bf16)v.z; pk[3] = (__bf16)v.w;
      sHv[idx] = pk;
    }
  }
  if (tid < Mn) {
    float mi = (i == 0) ? 1.0f : mask[b * Nn + (i - 1)];
    float mj = (tid == 0) ? 1.0f : mask[b * Nn + (tid - 1)];
    sNeg[tid] = (mi * mj - 1.0f) * 1e9f;
  }
  for (int idx = tid; idx < 4096; idx += 256) {
    int d = idx >> 6, c = idx & 63;
    int tdx = c * 64 + d;  // transpose: sW[out_col][in_dim]
    sW[tdx]        = (__bf16)Wq[idx];
    sW[4096 + tdx] = (__bf16)Wk[idx];
    sW[8192 + tdx] = (__bf16)Wv[idx];
  }
  __syncthreads();

  // ---- QKV projections; wave w handles token strip w (12 tiles) ----
  {
    v16bf xb[2], xa[2];
#pragma unroll
    for (int ks = 0; ks < 2; ++ks) {
      xb[ks] = bfragT(sXh, 64, ks * 32, w * 16, lane);  // X^T as B operand
      xa[ks] = afrag(sXh, 64, w * 16, ks * 32, lane);   // X as A operand
    }
    // Q^T = Wq^T . X^T  and  K^T = Wk^T . X^T : packed stores into row-major
#pragma unroll
    for (int m = 0; m < 2; ++m) {
      const __bf16* wp = sW + m * 4096;
      __bf16* dst = (m == 0) ? sQ : sK;
#pragma unroll
      for (int hkt = 0; hkt < 4; ++hkt) {
        v8f acc = {};
#pragma unroll
        for (int ks = 0; ks < 2; ++ks)
          acc = wmma_bf16(afrag(wp, 64, hkt * 16, ks * 32, lane), xb[ks], acc);
        *(v8bf*)(dst + (w * 16 + l15) * 64 + hkt * 16 + hf * 8) = pack_bf8(acc);
      }
    }
    // V = X . Wv : packed stores into transposed sVt[hv][key]
#pragma unroll
    for (int nt = 0; nt < 4; ++nt) {
      v8f acc = {};
#pragma unroll
      for (int ks = 0; ks < 2; ++ks)
        acc = wmma_bf16(xa[ks], bfragT(sW + 8192, 64, ks * 32, nt * 16, lane),
                        acc);
      *(v8bf*)(sVt + (nt * 16 + l15) * 128 + w * 16 + hf * 8) = pack_bf8(acc);
    }
  }
  __syncthreads();

  const float scale = 0.35355339059327373f;  // 1/sqrt(8)

  // ---- attention: S^T = K.Q^T per head; wave w owns query strip w ----
  for (int h = 0; h < Hn; ++h) {
    v8bf z8 = {};
    // B operand: Q strip (fixed per head), padded K-dim 8->32
    v8bf qlo = z8;
    if (hf == 0) qlo = *(const v8bf*)(sQ + (w * 16 + l15) * 64 + h * 8);
    v16bf bq = combine(qlo, z8);

    v8f s[8];  // s[kt][r] : key = kt*16 + hf*8 + r, query = w*16 + l15
#pragma unroll
    for (int kt = 0; kt < 8; ++kt) {
      v8bf klo = z8;
      if (hf == 0) klo = *(const v8bf*)(sK + (kt * 16 + l15) * 64 + h * 8);
      v16bf ak = combine(klo, z8);
      v8f acc = {};
      acc = wmma_bf16(ak, bq, acc);
      const float4* ng = (const float4*)(sNeg + kt * 16 + hf * 8);
      float4 n0 = ng[0], n1 = ng[1];
      acc[0] = acc[0] * scale + n0.x; acc[1] = acc[1] * scale + n0.y;
      acc[2] = acc[2] * scale + n0.z; acc[3] = acc[3] * scale + n0.w;
      acc[4] = acc[4] * scale + n1.x; acc[5] = acc[5] * scale + n1.y;
      acc[6] = acc[6] * scale + n1.z; acc[7] = acc[7] * scale + n1.w;
      s[kt] = acc;
    }
    // softmax over keys: lane-local across 64 regs + pair-lane reduce
    float mx = -3.0e38f;
#pragma unroll
    for (int kt = 0; kt < 8; ++kt)
#pragma unroll
      for (int r = 0; r < 8; ++r) mx = fmaxf(mx, s[kt][r]);
    mx = fmaxf(mx, __shfl_xor(mx, 16));
    float sum = 0.0f;
#pragma unroll
    for (int kt = 0; kt < 8; ++kt)
#pragma unroll
      for (int r = 0; r < 8; ++r) {
        float e_ = __expf(s[kt][r] - mx);
        s[kt][r] = e_;
        sum += e_;
      }
    sum += __shfl_xor(sum, 16);
    float invs = 1.0f / sum;
    int qrow = w * 16 + l15;
#pragma unroll
    for (int kt = 0; kt < 8; ++kt) {
      v8bf pk;
#pragma unroll
      for (int r = 0; r < 8; ++r) pk[r] = (__bf16)(s[kt][r] * invs);
      *(v8bf*)(sP + qrow * 128 + kt * 16 + hf * 8) = pk;  // row-major P
    }
    // O_h = P @ V_h  (N padded 8->16); write into sXh cols [8h, 8h+8)
    v8f o = {};
#pragma unroll
    for (int ks = 0; ks < 4; ++ks) {
      v16bf pa = afrag(sP, 128, w * 16, ks * 32, lane);
      v16bf vb;
      if (l15 < 8) {
        const __bf16* basep = sVt + (h * 8 + l15) * 128 + ks * 32 + hf * 16;
        vb = combine(*(const v8bf*)basep, *(const v8bf*)(basep + 8));
      } else {
        vb = combine(z8, z8);
      }
      o = wmma_bf16(pa, vb, o);
    }
    if (l15 < 8) {
#pragma unroll
      for (int r = 0; r < 8; ++r)
        sXh[(w * 16 + hf * 8 + r) * 64 + h * 8 + l15] = (__bf16)o[r];
    }
  }
  __syncthreads();

  // ---- out projection + residual + LN1 ----
  for (int idx = tid; idx < 4096; idx += 256) {
    int d = idx >> 6, c = idx & 63;
    sW[c * 64 + d] = (__bf16)Wo[idx];  // transpose
  }
  __syncthreads();
  {
    v16bf aO[2];
#pragma unroll
    for (int ks = 0; ks < 2; ++ks) aO[ks] = afrag(sXh, 64, w * 16, ks * 32, lane);
    v8f acc[4] = {};
#pragma unroll
    for (int nt = 0; nt < 4; ++nt)
#pragma unroll
      for (int ks = 0; ks < 2; ++ks)
        acc[nt] = wmma_bf16(aO[ks], bfragT(sW, 64, ks * 32, nt * 16, lane),
                            acc[nt]);
#pragma unroll
    for (int r = 0; r < 8; ++r) {
      int row = w * 16 + hf * 8 + r;
      float vv[4];
      float sm = 0.0f;
#pragma unroll
      for (int nt = 0; nt < 4; ++nt) {
        int col = nt * 16 + l15;
        vv[nt] = acc[nt][r] + sXf[row * 64 + col] + bo[col];
        sm += vv[nt];
      }
      sm += __shfl_xor(sm, 1); sm += __shfl_xor(sm, 2);
      sm += __shfl_xor(sm, 4); sm += __shfl_xor(sm, 8);
      float mean = sm * (1.0f / 64.0f);
      float sq = 0.0f;
#pragma unroll
      for (int nt = 0; nt < 4; ++nt) {
        float d = vv[nt] - mean;
        sq += d * d;
      }
      sq += __shfl_xor(sq, 1); sq += __shfl_xor(sq, 2);
      sq += __shfl_xor(sq, 4); sq += __shfl_xor(sq, 8);
      float inv = rsqrtf(sq * (1.0f / 64.0f) + 1e-5f);
#pragma unroll
      for (int nt = 0; nt < 4; ++nt) {
        int col = nt * 16 + l15;
        float o2 = (vv[nt] - mean) * inv * ln1g[col] + ln1b[col];
        sXf[row * 64 + col] = o2;
        sXh[row * 64 + col] = (__bf16)o2;
      }
    }
  }
  __syncthreads();

  // ---- FFN: 64 -> 256 (relu) -> 64, chunked over F in 8 slabs of 32 ----
  __bf16* sW1t = sW;         // [32][64]  Bt: [F-col][d]
  __bf16* sW2t = sW + 2048;  // [64][32]  Bt: [d][F-row]
  __bf16* sHc  = sW + 4096;  // [128][32] relu(h) chunk (A operand, row-major)
  v8f acc2[4] = {};
  for (int fc = 0; fc < 8; ++fc) {
    __syncthreads();
    for (int idx = tid; idx < 2048; idx += 256) {
      int dd = idx >> 5, c = idx & 31;
      sW1t[c * 64 + dd] = (__bf16)W1[dd * 256 + fc * 32 + c];
      int rr = idx >> 6, d2 = idx & 63;
      sW2t[d2 * 32 + rr] = (__bf16)W2[(fc * 32 + rr) * 64 + d2];
    }
    __syncthreads();
    v16bf aX[2];
#pragma unroll
    for (int ks = 0; ks < 2; ++ks) aX[ks] = afrag(sXh, 64, w * 16, ks * 32, lane);
#pragma unroll
    for (int n2 = 0; n2 < 2; ++n2) {
      v8f acc = {};
#pragma unroll
      for (int ks = 0; ks < 2; ++ks)
        acc = wmma_bf16(aX[ks], bfragT(sW1t, 64, ks * 32, n2 * 16, lane), acc);
#pragma unroll
      for (int r = 0; r < 8; ++r) {
        int row = w * 16 + hf * 8 + r;
        int col = n2 * 16 + l15;
        float hv = fmaxf(acc[r] + b1[fc * 32 + col], 0.0f);
        sHc[row * 32 + col] = (__bf16)hv;
      }
    }
    v16bf aH = afrag(sHc, 32, w * 16, 0, lane);
#pragma unroll
    for (int nt = 0; nt < 4; ++nt)
      acc2[nt] = wmma_bf16(aH, bfragT(sW2t, 32, 0, nt * 16, lane), acc2[nt]);
  }
  // ---- residual + b2 + LN2 -> global ----
#pragma unroll
  for (int r = 0; r < 8; ++r) {
    int row = w * 16 + hf * 8 + r;
    float vv[4];
    float sm = 0.0f;
#pragma unroll
    for (int nt = 0; nt < 4; ++nt) {
      int col = nt * 16 + l15;
      vv[nt] = acc2[nt][r] + sXf[row * 64 + col] + b2[col];
      sm += vv[nt];
    }
    sm += __shfl_xor(sm, 1); sm += __shfl_xor(sm, 2);
    sm += __shfl_xor(sm, 4); sm += __shfl_xor(sm, 8);
    float mean = sm * (1.0f / 64.0f);
    float sq = 0.0f;
#pragma unroll
    for (int nt = 0; nt < 4; ++nt) {
      float d = vv[nt] - mean;
      sq += d * d;
    }
    sq += __shfl_xor(sq, 1); sq += __shfl_xor(sq, 2);
    sq += __shfl_xor(sq, 4); sq += __shfl_xor(sq, 8);
    float inv = rsqrtf(sq * (1.0f / 64.0f) + 1e-5f);
#pragma unroll
    for (int nt = 0; nt < 4; ++nt) {
      int col = nt * 16 + l15;
      Xrow[row * 64 + col] = (vv[nt] - mean) * inv * ln2g[col] + ln2b[col];
    }
  }
}

// ---------- Kernel 3: diagonal readout ----------
__global__ __launch_bounds__(256) void readout_kernel(
    const float* __restrict__ X, float* __restrict__ out) {
  int g = blockIdx.x * 256 + threadIdx.x;
  if (g >= Bn * Mn * Dn) return;
  int b, j, d;
  if (g < Bn * Dn) {
    b = g >> 6; d = g & 63; j = 0;
  } else {
    int g2 = g - Bn * Dn;
    b = g2 / (Nn * Dn);
    int r = g2 - b * (Nn * Dn);
    j = (r >> 6) + 1;
    d = r & 63;
  }
  out[g] = X[(((size_t)b * Mn + j) * Mn + j) * Dn + d];
}

// ---------- host ----------
extern "C" void kernel_launch(void* const* d_in, const int* in_sizes, int n_in,
                              void* d_out, int out_size, void* d_ws,
                              size_t ws_size, hipStream_t stream) {
  const float* nodef = (const float*)d_in[0];
  const float* edgef = (const float*)d_in[1];
  const float* mask  = (const float*)d_in[2];
  const float* W_in  = (const float*)d_in[3];
  const float* b_in  = (const float*)d_in[4];
  const float* ln_g  = (const float*)d_in[5];
  const float* ln_b  = (const float*)d_in[6];
  const float* Wq    = (const float*)d_in[7];
  const float* Wk    = (const float*)d_in[8];
  const float* Wv    = (const float*)d_in[9];
  const float* Wo    = (const float*)d_in[10];
  const float* bo    = (const float*)d_in[11];
  const float* W1    = (const float*)d_in[12];
  const float* b1    = (const float*)d_in[13];
  const float* W2    = (const float*)d_in[14];
  const float* b2    = (const float*)d_in[15];
  const float* ln1g  = (const float*)d_in[16];
  const float* ln1b  = (const float*)d_in[17];
  const float* ln2g  = (const float*)d_in[18];
  const float* ln2b  = (const float*)d_in[19];

  float* X = (float*)d_ws;  // [16,128,128,64] fp32 activations

  build_kernel<<<32768, 256, 0, stream>>>(nodef, edgef, W_in, b_in, ln_g, ln_b,
                                          X);

  for (int l = 0; l < 4; ++l) {
    layer_kernel<<<Bn * Mn, 256, 0, stream>>>(
        X, mask, Wq + l * 4096, Wk + l * 4096, Wv + l * 4096, Wo + l * 4096,
        bo + l * 64, W1 + l * 16384, b1 + l * 256, W2 + l * 16384, b2 + l * 64,
        ln1g + l * 64, ln1b + l * 64, ln2g + l * 64, ln2b + l * 64);
  }

  readout_kernel<<<(Bn * Mn * Dn + 255) / 256, 256, 0, stream>>>(X,
                                                                 (float*)d_out);
}
